// Code2LoRAGRU_71597104824555
// MI455X (gfx1250) — compile-verified, matrix operand order checked by reading, not run on hardware
//
#include <hip/hip_runtime.h>
#include <math.h>

// ---------------- problem dims ----------------
#define BB   4
#define TT   512
#define DINN 1536
#define HHH  1024
#define HH3  3072
#define THD  512
#define TFFD 2048
#define LLL  32
#define MMM  2
#define NNN  16
#define RRR  16
#define DDD  4096
#define NCOE (LLL*MMM*NNN*2)   /* 2048 head outputs */
#define ROWS1 (BB*TT)          /* 2048 */
#define GRU_BLOCKS 8

typedef __attribute__((ext_vector_type(16))) __bf16 v16bf;
typedef __attribute__((ext_vector_type(8)))  float  v8f;

// Load one 16x32 bf16 WMMA operand fragment for this lane.
// base points at (row-or-col)*K + kk. Per ISA 7.12.2: lane half h needs
// k in {8h..8h+7} and {16+8h..16+8h+7} -> two 16-byte chunks.
__device__ __forceinline__ v16bf frag_load(const __bf16* base, int half) {
  union { uint4 u[2]; v16bf v; } r;
  r.u[0] = *reinterpret_cast<const uint4*>(base + half * 8);
  r.u[1] = *reinterpret_cast<const uint4*>(base + 16 + half * 8);
  return r.v;
}

// ---------------- conversion kernels ----------------
__global__ void k_cvt_bf16(const float* __restrict__ in, __bf16* __restrict__ out, int n) {
  int i = blockIdx.x * blockDim.x + threadIdx.x;
  if (i < n) out[i] = (__bf16)in[i];
}

// in[rows,cols] (row-major f32) -> out[cols,rows] (row-major bf16) == W^T
__global__ void k_cvt_t_bf16(const float* __restrict__ in, __bf16* __restrict__ out,
                             int rows, int cols) {
  int i = blockIdx.x * blockDim.x + threadIdx.x;
  if (i < rows * cols) {
    int r = i / cols, c = i % cols;
    out[(size_t)c * rows + r] = (__bf16)in[i];
  }
}

__global__ void k_init_state(float* h0f, float* h1f, __bf16* h0b, __bf16* h1b,
                             unsigned* ctr) {
  int i = blockIdx.x * blockDim.x + threadIdx.x;
  if (i < 16 * HHH) {
    h0f[i] = 0.f; h1f[i] = 0.f;
    h0b[i] = (__bf16)0.f; h1b[i] = (__bf16)0.f;
  }
  if (i == 0) *ctr = 0u;
}

// ---------------- generic WMMA GEMM, MT M-tiles per wave ----------------
// C[M,N] (f32) = A[M,K](bf16 row-major) x BT[N,K](bf16 row-major, i.e. B^T) + bias[N]
// Each wave computes MT vertically-stacked 16x16 C tiles sharing one B fragment
// per k-step (MT-fold reuse of the streamed B operand -> MT x less L2 traffic).
template <int MT>
__global__ __launch_bounds__(256) void k_gemm_bf16(
    const __bf16* __restrict__ A, const __bf16* __restrict__ BT,
    const float* __restrict__ bias, float* __restrict__ C,
    int M, int N, int K) {
  int wave = threadIdx.x >> 5;
  int lane = threadIdx.x & 31;
  int half = lane >> 4, l16 = lane & 15;
  int ntn = N >> 4;
  int mgroups = (M >> 4) / MT;
  int tile = blockIdx.x * 8 + wave;
  int total = mgroups * ntn;
  if (tile >= total) return;                 // wave-uniform: EXEC stays all-ones
  int tg = tile / ntn, tn = tile - tg * ntn;
  int tm0 = tg * MT;

  const __bf16* arow[MT];
#pragma unroll
  for (int mt = 0; mt < MT; ++mt)
    arow[mt] = A + (size_t)((tm0 + mt) * 16 + l16) * K;
  const __bf16* brow = BT + (size_t)(tn * 16 + l16) * K;

  v8f acc[MT];
#pragma unroll
  for (int mt = 0; mt < MT; ++mt) acc[mt] = (v8f){};

  for (int kk = 0; kk < K; kk += 32) {
    __builtin_prefetch((const void*)(brow + kk + 256), 0, 1);  // global_prefetch_b8
    v16bf b = frag_load(brow + kk, half);
#pragma unroll
    for (int mt = 0; mt < MT; ++mt) {
      v16bf a = frag_load(arow[mt] + kk, half);
      acc[mt] = __builtin_amdgcn_wmma_f32_16x16x32_bf16(false, a, false, b,
                                                        (short)0, acc[mt], false, false);
    }
  }
  float bv = bias[tn * 16 + l16];
#pragma unroll
  for (int mt = 0; mt < MT; ++mt) {
    float* cbase = C + (size_t)((tm0 + mt) * 16) * N + tn * 16 + l16;
#pragma unroll
    for (int r = 0; r < 8; ++r)
      cbase[(size_t)(r + half * 8) * N] = acc[mt][r] + bv;
  }
}

// ---------------- layernorm (f32 in -> bf16 out), one block per row ----------------
__global__ __launch_bounds__(256) void k_layernorm(
    const float* __restrict__ in, const float* __restrict__ g,
    const float* __restrict__ b, __bf16* __restrict__ out, int n) {
  __shared__ float ssum[256], ssq[256];
  int row = blockIdx.x, tid = threadIdx.x;
  const float* x = in + (size_t)row * n;
  float s = 0.f, q = 0.f;
  for (int i = tid; i < n; i += 256) { float v = x[i]; s += v; q += v * v; }
  ssum[tid] = s; ssq[tid] = q; __syncthreads();
  for (int off = 128; off > 0; off >>= 1) {
    if (tid < off) { ssum[tid] += ssum[tid + off]; ssq[tid] += ssq[tid + off]; }
    __syncthreads();
  }
  float mean = ssum[0] / n;
  float var = ssq[0] / n - mean * mean;
  float inv = rsqrtf(var + 1e-5f);
  __bf16* o = out + (size_t)row * n;
  for (int i = tid; i < n; i += 256)
    o[i] = (__bf16)(((x[i] - mean) * inv) * g[i] + b[i]);
}

// ---------------- elementwise ----------------
__global__ void k_gelu(const float* __restrict__ in, float* __restrict__ outf,
                       __bf16* __restrict__ outb, int n) {
  int i = blockIdx.x * blockDim.x + threadIdx.x;
  if (i < n) {
    float v = in[i];
    float ge = 0.5f * v * (1.f + erff(v * 0.7071067811865476f));
    if (outf) outf[i] = ge;
    outb[i] = (__bf16)ge;
  }
}

__global__ void k_add_bf16(const float* __restrict__ a, const float* __restrict__ c,
                           __bf16* __restrict__ outb, int n) {
  int i = blockIdx.x * blockDim.x + threadIdx.x;
  if (i < n) outb[i] = (__bf16)(a[i] + c[i]);
}

// ---------------- GRU scan: persistent grid, grid-wide barrier per step ----------------
// 8 blocks x 8 waves = 64 waves; wave gw owns output columns [16gw,16gw+16) of the
// r/z/n gates (rows = batch, padded 4->16). h ping-pongs between buf0/buf1.
// Whh (bf16, 6.3MB) stays L2-resident across the 512 steps.
__global__ __launch_bounds__(256) void k_gru_scan(
    const __bf16* __restrict__ WhhT,   // gru_Whh as-is: [3H, H] row-major == B^T
    const float* __restrict__ xg,      // [B*T, 3H]
    const float* __restrict__ bhh,     // [3H]
    const int* __restrict__ lengths,   // [B]
    float* h0f, float* h1f, __bf16* h0b, __bf16* h1b,
    unsigned* __restrict__ ctr) {
  int wave = threadIdx.x >> 5, lane = threadIdx.x & 31;
  int half = lane >> 4, l16 = lane & 15;
  int gw = blockIdx.x * 8 + wave;      // 0..63
  int c0 = gw * 16;
  int col = c0 + l16;

  const __bf16* brr = WhhT + (size_t)(c0 + l16) * HHH;
  const __bf16* bzr = WhhT + (size_t)(HHH + c0 + l16) * HHH;
  const __bf16* bnr = WhhT + (size_t)(2 * HHH + c0 + l16) * HHH;
  float bhr = bhh[col], bhz = bhh[HHH + col], bhn = bhh[2 * HHH + col];

  for (int t = 0; t < TT; ++t) {
    const float*  hfR = (t & 1) ? h1f : h0f;
    float*        hfW = (t & 1) ? h0f : h1f;
    const __bf16* hbR = (t & 1) ? h1b : h0b;
    __bf16*       hbW = (t & 1) ? h0b : h1b;

    v8f ar = {}, az = {}, an = {};
    const __bf16* hrow = hbR + (size_t)l16 * HHH;
    for (int kk = 0; kk < HHH; kk += 32) {
      v16bf a  = frag_load(hrow + kk, half);
      v16bf wr = frag_load(brr + kk, half);
      v16bf wz = frag_load(bzr + kk, half);
      v16bf wn = frag_load(bnr + kk, half);
      ar = __builtin_amdgcn_wmma_f32_16x16x32_bf16(false, a, false, wr, (short)0, ar, false, false);
      az = __builtin_amdgcn_wmma_f32_16x16x32_bf16(false, a, false, wz, (short)0, az, false, false);
      an = __builtin_amdgcn_wmma_f32_16x16x32_bf16(false, a, false, wn, (short)0, an, false, false);
    }
#pragma unroll
    for (int r = 0; r < 8; ++r) {
      int b = r + half * 8;              // C row = batch index
      if (b < BB) {
        size_t xrow = ((size_t)b * TT + t) * HH3;
        float xwr = xg[xrow + col];
        float xwz = xg[xrow + HHH + col];
        float xwn = xg[xrow + 2 * HHH + col];
        float hhr = ar[r] + bhr, hhz = az[r] + bhz, hhn = an[r] + bhn;
        float rg = 1.f / (1.f + __expf(-(xwr + hhr)));
        float zg = 1.f / (1.f + __expf(-(xwz + hhz)));
        float ng = tanhf(xwn + rg * hhn);
        float hp = hfR[(size_t)b * HHH + col];
        int len = lengths[b]; if (len < 1) len = 1;
        float hn = (t < len) ? ((1.f - zg) * ng + zg * hp) : hp;
        hfW[(size_t)b * HHH + col] = hn;
        hbW[(size_t)b * HHH + col] = (__bf16)hn;
      }
    }
    // grid-wide barrier: make this step's h visible before anyone starts step t+1
    __threadfence();
    __syncthreads();
    if (threadIdx.x == 0) {
      __hip_atomic_fetch_add(ctr, 1u, __ATOMIC_ACQ_REL, __HIP_MEMORY_SCOPE_AGENT);
      unsigned target = (unsigned)GRU_BLOCKS * (unsigned)(t + 1);
      while (__hip_atomic_load(ctr, __ATOMIC_ACQUIRE, __HIP_MEMORY_SCOPE_AGENT) < target)
        __builtin_amdgcn_s_sleep(1);
    }
    __syncthreads();
  }
}

// ---------------- LoRA basis expansion ----------------
// out[b,l,s,r,d]: s in 0..1 -> A modules (m=s), s in 2..3 -> B^T modules (m=s-2)
__global__ __launch_bounds__(256) void k_expand(
    const float* __restrict__ coef,   // [16, 2048], rows 0..3 valid
    const float* __restrict__ Abas,   // [M,N,R,D]
    const float* __restrict__ Bbas,   // [M,N,D,R]
    float* __restrict__ out) {
  int blk = blockIdx.x;
  int r = blk & 15; blk >>= 4;
  int s = blk & 3;  blk >>= 2;
  int l = blk % LLL;
  int b = blk / LLL;
  int m = s & 1;
  bool isA = (s < 2);
  int comp = isA ? 0 : 1;

  float cf[NNN];
  const float* crow = coef + (size_t)b * NCOE + ((l * MMM + m) * NNN) * 2 + comp;
#pragma unroll
  for (int n = 0; n < NNN; ++n) cf[n] = crow[n * 2];

  float* orow = out + ((((size_t)(b * LLL + l) * 4 + s) * RRR + r) * DDD);
  if (isA) {
    for (int d = threadIdx.x * 4; d < DDD; d += 256 * 4) {
      float4 acc = {0.f, 0.f, 0.f, 0.f};
#pragma unroll
      for (int n = 0; n < NNN; ++n) {
        const float4 v = *reinterpret_cast<const float4*>(
            Abas + (((size_t)(m * NNN + n) * RRR + r) * DDD + d));
        acc.x += cf[n] * v.x; acc.y += cf[n] * v.y;
        acc.z += cf[n] * v.z; acc.w += cf[n] * v.w;
      }
      *reinterpret_cast<float4*>(orow + d) = acc;
    }
  } else {
    for (int d = threadIdx.x; d < DDD; d += 256) {
      float acc = 0.f;
#pragma unroll
      for (int n = 0; n < NNN; ++n)
        acc += cf[n] * Bbas[((size_t)(m * NNN + n) * DDD + d) * RRR + r];
      orow[d] = acc;
    }
  }
}

// ---------------- host launch ----------------
extern "C" void kernel_launch(void* const* d_in, const int* in_sizes, int n_in,
                              void* d_out, int out_size, void* d_ws, size_t ws_size,
                              hipStream_t stream) {
  (void)in_sizes; (void)n_in; (void)out_size; (void)ws_size;
  const float* fe      = (const float*)d_in[0];
  const int*   lengths = (const int*)  d_in[1];
  const float* Win     = (const float*)d_in[2];
  const float* bin     = (const float*)d_in[3];
  const float* g_in    = (const float*)d_in[4];
  const float* b_in    = (const float*)d_in[5];
  const float* Wih     = (const float*)d_in[6];
  const float* Whh     = (const float*)d_in[7];
  const float* bih     = (const float*)d_in[8];
  const float* bhh     = (const float*)d_in[9];
  const float* g_out   = (const float*)d_in[10];
  const float* b_out   = (const float*)d_in[11];
  const float* Wtr     = (const float*)d_in[12];
  const float* btr     = (const float*)d_in[13];
  const float* g_mlp   = (const float*)d_in[14];
  const float* b_mlp   = (const float*)d_in[15];
  const float* W1      = (const float*)d_in[16];
  const float* b1      = (const float*)d_in[17];
  const float* W2      = (const float*)d_in[18];
  const float* b2      = (const float*)d_in[19];
  const float* Whd     = (const float*)d_in[20];
  const float* bhd     = (const float*)d_in[21];
  const float* Abas    = (const float*)d_in[22];
  const float* Bbas    = (const float*)d_in[23];
  float* out = (float*)d_out;

  char* ws = (char*)d_ws;
  size_t off = 0;
  auto alloc = [&](size_t bytes) -> void* {
    void* p = ws + off;
    off += (bytes + 255) & ~(size_t)255;
    return p;
  };

  __bf16* fe_bf   = (__bf16*)alloc((size_t)ROWS1 * DINN * 2);
  __bf16* WinT_bf = (__bf16*)alloc((size_t)HHH * DINN * 2);
  __bf16* Wih_bf  = (__bf16*)alloc((size_t)HH3 * HHH * 2);
  __bf16* Whh_bf  = (__bf16*)alloc((size_t)HH3 * HHH * 2);
  __bf16* WtrT_bf = (__bf16*)alloc((size_t)THD * HHH * 2);
  __bf16* W1T_bf  = (__bf16*)alloc((size_t)TFFD * THD * 2);
  __bf16* W2T_bf  = (__bf16*)alloc((size_t)THD * TFFD * 2);
  __bf16* WhdT_bf = (__bf16*)alloc((size_t)NCOE * THD * 2);
  float*  pre     = (float*) alloc((size_t)ROWS1 * HHH * 4);
  __bf16* x_bf    = (__bf16*)alloc((size_t)ROWS1 * HHH * 2);
  float*  xg      = (float*) alloc((size_t)ROWS1 * HH3 * 4);
  float*  h0f     = (float*) alloc(16 * HHH * 4);
  float*  h1f     = (float*) alloc(16 * HHH * 4);
  __bf16* h0b     = (__bf16*)alloc(16 * HHH * 2);
  __bf16* h1b     = (__bf16*)alloc(16 * HHH * 2);
  __bf16* hln_bf  = (__bf16*)alloc(16 * HHH * 2);
  float*  t0      = (float*) alloc(16 * THD * 4);
  float*  tact    = (float*) alloc(16 * THD * 4);
  __bf16* t_bf    = (__bf16*)alloc(16 * THD * 2);
  __bf16* tln_bf  = (__bf16*)alloc(16 * THD * 2);
  float*  u0      = (float*) alloc(16 * TFFD * 4);
  __bf16* u_bf    = (__bf16*)alloc(16 * TFFD * 2);
  float*  v0      = (float*) alloc(16 * THD * 4);
  __bf16* t2_bf   = (__bf16*)alloc(16 * THD * 2);
  float*  coef    = (float*) alloc(16 * NCOE * 4);
  unsigned* ctr   = (unsigned*)alloc(256);

  auto cdiv = [](int a, int b) { return (a + b - 1) / b; };
  auto gemm = [&](const __bf16* A, const __bf16* BT, const float* bias, float* C,
                  int M, int N, int K) {
    int mtiles = M / 16, ntiles = N / 16;
    if (mtiles % 4 == 0) {
      int tiles = (mtiles / 4) * ntiles;
      k_gemm_bf16<4><<<cdiv(tiles, 8), 256, 0, stream>>>(A, BT, bias, C, M, N, K);
    } else {
      int tiles = mtiles * ntiles;
      k_gemm_bf16<1><<<cdiv(tiles, 8), 256, 0, stream>>>(A, BT, bias, C, M, N, K);
    }
  };

  // one-time (per call) weight/activation conversions to bf16
  k_cvt_bf16 <<<cdiv(ROWS1 * DINN, 256), 256, 0, stream>>>(fe, fe_bf, ROWS1 * DINN);
  k_cvt_t_bf16<<<cdiv(DINN * HHH, 256), 256, 0, stream>>>(Win, WinT_bf, DINN, HHH);
  k_cvt_bf16 <<<cdiv(HH3 * HHH, 256), 256, 0, stream>>>(Wih, Wih_bf, HH3 * HHH);
  k_cvt_bf16 <<<cdiv(HH3 * HHH, 256), 256, 0, stream>>>(Whh, Whh_bf, HH3 * HHH);
  k_cvt_t_bf16<<<cdiv(HHH * THD, 256), 256, 0, stream>>>(Wtr, WtrT_bf, HHH, THD);
  k_cvt_t_bf16<<<cdiv(THD * TFFD, 256), 256, 0, stream>>>(W1, W1T_bf, THD, TFFD);
  k_cvt_t_bf16<<<cdiv(TFFD * THD, 256), 256, 0, stream>>>(W2, W2T_bf, TFFD, THD);
  k_cvt_t_bf16<<<cdiv(THD * NCOE, 256), 256, 0, stream>>>(Whd, WhdT_bf, THD, NCOE);
  k_init_state<<<64, 256, 0, stream>>>(h0f, h1f, h0b, h1b, ctr);

  // input projection + LN
  gemm(fe_bf, WinT_bf, bin, pre, ROWS1, HHH, DINN);
  k_layernorm<<<ROWS1, 256, 0, stream>>>(pre, g_in, b_in, x_bf, HHH);

  // precompute input gates xg = x @ Wih^T + bih
  gemm(x_bf, Wih_bf, bih, xg, ROWS1, HH3, HHH);

  // sequential GRU scan (persistent kernel, grid barrier per step)
  k_gru_scan<<<GRU_BLOCKS, 256, 0, stream>>>(Whh_bf, xg, bhh, lengths,
                                             h0f, h1f, h0b, h1b, ctr);

  // output LN of h_T (final h lands in buffer 0)
  k_layernorm<<<BB, 256, 0, stream>>>(h0f, g_out, b_out, hln_bf, HHH);

  // trunk: Linear + GELU, residual MLP block, head
  gemm(hln_bf, WtrT_bf, btr, t0, 16, THD, HHH);
  k_gelu<<<cdiv(16 * THD, 256), 256, 0, stream>>>(t0, tact, t_bf, 16 * THD);
  k_layernorm<<<BB, 256, 0, stream>>>(tact, g_mlp, b_mlp, tln_bf, THD);
  gemm(tln_bf, W1T_bf, b1, u0, 16, TFFD, THD);
  k_gelu<<<cdiv(16 * TFFD, 256), 256, 0, stream>>>(u0, nullptr, u_bf, 16 * TFFD);
  gemm(u_bf, W2T_bf, b2, v0, 16, THD, TFFD);
  k_add_bf16<<<cdiv(16 * THD, 256), 256, 0, stream>>>(tact, v0, t2_bf, 16 * THD);
  gemm(t2_bf, WhdT_bf, bhd, coef, 16, NCOE, THD);

  // LoRA basis combination (HBM-write bound)
  k_expand<<<BB * LLL * 4 * RRR, 256, 0, stream>>>(coef, Abas, Bbas, out);
}
